// MonolithicDecoderLayer_39092792328759
// MI455X (gfx1250) — compile-verified, matrix operand order checked by reading, not run on hardware
//
#include <hip/hip_runtime.h>
#include <hip/hip_bf16.h>
#include <math.h>

// ---------------- model constants ----------------
#define T_ 2048
#define HID_ 2048
#define QL_ 512
#define KVL_ 512
#define DR_ 64
#define DN_ 128
#define DV_ 128
#define NH_ 16
#define IDXH_ 16
#define IDXD_ 128
#define TOPK_ 512
#define INTER_ 8192
#define STEP1N_ (QL_ + KVL_ + DR_ + IDXD_ + IDXH_)   // 1232
#define QFN_ (NH_ * (DN_ + DR_))                      // 3072
#define AW_ (NH_ * (KVL_ + DR_))                      // 9216 (concatenated q_lat|q_pe per head)
#define A576_ (KVL_ + DR_)                            // 576

typedef __attribute__((ext_vector_type(16))) __bf16 v16bf;
typedef __attribute__((ext_vector_type(8)))  float  v8f;
typedef __attribute__((ext_vector_type(4)))  unsigned int v4u;
typedef __attribute__((ext_vector_type(8)))  unsigned int v8u;

enum { EPI_F32 = 0, EPI_BF16 = 1, EPI_IDX = 2, EPI_SCORES = 3 };

// ---------------- bf16 WMMA GEMM: C = A (MxK, row major) * B^T (B is NxK row major) ----------------
#define BM 128
#define BN 128
#define BK 32

// Issue a TDM 2D tile load (rows x BK bf16 elements, row-major) into LDS.
// D# per CDNA5 ISA ch.8: group0 = {count/flags, lds_addr, global_addr, type=2},
// group1 = {mask/data_size, tensor dims (remaining extents for OOB zero-fill),
//           tile dims, dim0 stride}. Tracked by TENSORcnt.
__device__ __forceinline__ void tdm_load_tile(const __bf16* gsrc, long ldElems,
                                              int rowsLeft, int kLeft, void* ldsDst)
{
  const unsigned long long ga = (unsigned long long)(uintptr_t)gsrc;
  const unsigned lds = (unsigned)(uintptr_t)ldsDst;   // low 32 bits of flat addr = LDS offset
  const unsigned td0 = (unsigned)kLeft;               // remaining row extent (data_size units)
  const unsigned td1 = (unsigned)rowsLeft;            // remaining rows
  const unsigned long long st = (unsigned long long)ldElems;

  v4u g0;
  g0[0] = 1u;                                           // count=1, user descriptor
  g0[1] = lds;                                          // lds_addr (bytes)
  g0[2] = (unsigned)(ga & 0xFFFFFFFFu);                 // global_addr[31:0]
  g0[3] = (unsigned)((ga >> 32) & 0x01FFFFFFu) | (2u << 30);  // addr[56:32] | type=2

  v8u g1;
  g1[0] = (1u << 16);                                   // workgroup_mask=0, data_size=1 (2 bytes)
  g1[1] = (td0 & 0xFFFFu) << 16;                        // tensor_dim0[15:0]
  g1[2] = (td0 >> 16) | ((td1 & 0xFFFFu) << 16);        // tensor_dim0[31:16] | tensor_dim1[15:0]
  g1[3] = (td1 >> 16) | ((unsigned)BK << 16);           // tensor_dim1[31:16] | tile_dim0=BK
  g1[4] = (unsigned)BM;                                 // tile_dim1=BM, tile_dim2=0
  g1[5] = (unsigned)(st & 0xFFFFFFFFu);                 // tensor_dim0_stride[31:0]
  g1[6] = (unsigned)((st >> 32) & 0xFFFFu);             // stride[47:32] | dim1_stride lo = 0
  g1[7] = 0u;

  asm volatile("tensor_load_to_lds %0, %1" :: "s"(g0), "s"(g1) : "memory");
}

__global__ __launch_bounds__(256) void gemm_bf16_wmma(
    const __bf16* __restrict__ A, int lda, long sA,
    const __bf16* __restrict__ B, int ldb, long sB,
    void* __restrict__ Cv, int ldc, long sC,
    int M, int N, int K,
    float alpha, int epi,
    const float* __restrict__ aux0, const float* __restrict__ aux1,
    int auxStride, int beta)
{
  __shared__ __bf16 As[2][BM][BK];
  __shared__ __bf16 Bs[2][BN][BK];

  const int z = blockIdx.z;
  A += (size_t)z * sA;
  B += (size_t)z * sB;

  const int tileM = blockIdx.y * BM;
  const int tileN = blockIdx.x * BN;
  const int tid   = threadIdx.x;
  const int lane  = tid & 31;
  const int wave  = tid >> 5;
  const int wm    = (wave >> 1) * 32;   // wave row offset within block tile
  const int wn    = (wave & 1) * 64;    // wave col offset within block tile
  const int hlf   = lane >> 4;          // half-wave select (K split per ISA layout)
  const int lm    = lane & 15;

  const bool issuer = (wave == 0);      // one wave drives the Tensor Data Mover
  const int rowsA = M - tileM;
  const int rowsB = N - tileN;

  v8f acc[2][4] = {};

  const int nK = K / BK;

  // Prime stage 0 via TDM.
  if (issuer) {
    tdm_load_tile(A + (size_t)tileM * lda, lda, rowsA, K, &As[0][0][0]);
    tdm_load_tile(B + (size_t)tileN * ldb, ldb, rowsB, K, &Bs[0][0][0]);
    __builtin_amdgcn_s_wait_tensorcnt(0);
  }
  __syncthreads();

  for (int kt = 0; kt < nK; ++kt) {
    const int cur = kt & 1;
    if (issuer && (kt + 1 < nK)) {  // DMA next K-step tiles into the other buffer
      tdm_load_tile(A + (size_t)tileM * lda + (size_t)(kt + 1) * BK, lda,
                    rowsA, K - (kt + 1) * BK, &As[cur ^ 1][0][0]);
      tdm_load_tile(B + (size_t)tileN * ldb + (size_t)(kt + 1) * BK, ldb,
                    rowsB, K - (kt + 1) * BK, &Bs[cur ^ 1][0][0]);
    }

    // Gather WMMA fragments from LDS per CDNA5 16-bit A/B layout:
    // element e -> K index: ((p<4)?0:16) + (p&3)*2 + (e&1) + halfwave*8, p = e>>1
    v16bf af[2], bfg[4];
#pragma unroll
    for (int e = 0; e < 16; ++e) {
      const int p = e >> 1, o = e & 1;
      const int k = ((p < 4) ? 0 : 16) + ((p & 3) << 1) + o + hlf * 8;
      af[0][e]  = As[cur][wm + lm][k];
      af[1][e]  = As[cur][wm + 16 + lm][k];
      bfg[0][e] = Bs[cur][wn + lm][k];
      bfg[1][e] = Bs[cur][wn + 16 + lm][k];
      bfg[2][e] = Bs[cur][wn + 32 + lm][k];
      bfg[3][e] = Bs[cur][wn + 48 + lm][k];
    }
#pragma unroll
    for (int mi = 0; mi < 2; ++mi)
#pragma unroll
      for (int ni = 0; ni < 4; ++ni)
        acc[mi][ni] = __builtin_amdgcn_wmma_f32_16x16x32_bf16(
            false, af[mi], false, bfg[ni], (short)0, acc[mi][ni], false, false);

    if (issuer) __builtin_amdgcn_s_wait_tensorcnt(0);
    __syncthreads();
  }

  // Epilogue. C layout (ISA 7.12.2): VGPR i, lanes0-15 -> M=i, lanes16-31 -> M=i+8; N = lane%16.
#pragma unroll
  for (int mi = 0; mi < 2; ++mi) {
#pragma unroll
    for (int ni = 0; ni < 4; ++ni) {
      const int c = tileN + wn + ni * 16 + lm;
      if (c >= N) continue;
#pragma unroll
      for (int i = 0; i < 8; ++i) {
        const int r = tileM + wm + mi * 16 + hlf * 8 + i;
        if (r >= M) continue;
        const float v = acc[mi][ni][i] * alpha;
        const size_t idx = (size_t)r * ldc + c;
        if (epi == EPI_F32) {
          ((float*)Cv + (size_t)z * sC)[idx] = v;
        } else if (epi == EPI_BF16) {
          ((__bf16*)Cv + (size_t)z * sC)[idx] = (__bf16)v;
        } else if (epi == EPI_IDX) {
          const float val = fmaxf(v, 0.f) * aux0[(size_t)r * auxStride];
          float* o = (float*)Cv;
          o[idx] = beta ? (o[idx] + val) : val;
        } else { // EPI_SCORES
          const bool sel = (c <= r) && (aux0[(size_t)r * auxStride + c] >= aux1[r]);
          ((float*)Cv)[idx] = sel ? v : -3.0e38f;
        }
      }
    }
  }
}

// ---------------- fused residual add + rmsnorm (writes sum f32 and normalized bf16) ----------------
__global__ __launch_bounds__(256) void add_rmsnorm_kernel(
    const float* __restrict__ a, const float* __restrict__ b, const float* __restrict__ w,
    float* __restrict__ rout, __bf16* __restrict__ hout, int cols)
{
  const int row = blockIdx.x;
  a += (size_t)row * cols; b += (size_t)row * cols;
  rout += (size_t)row * cols; hout += (size_t)row * cols;
  __shared__ float sb[256];
  float ss = 0.f;
  for (int c = threadIdx.x; c < cols; c += 256) { float v = a[c] + b[c]; rout[c] = v; ss += v * v; }
  sb[threadIdx.x] = ss; __syncthreads();
  for (int s = 128; s > 0; s >>= 1) { if (threadIdx.x < s) sb[threadIdx.x] += sb[threadIdx.x + s]; __syncthreads(); }
  const float inv = rsqrtf(sb[0] / cols + 1e-6f);
  for (int c = threadIdx.x; c < cols; c += 256) hout[c] = (__bf16)(rout[c] * inv * w[c]);
}

// ---------------- rmsnorm on strided f32 src -> bf16 (and optional f32) ----------------
__global__ __launch_bounds__(256) void rmsnorm_kernel(
    const float* __restrict__ src, int sld, const float* __restrict__ w,
    __bf16* __restrict__ dstb, int dldb, float* __restrict__ dstf, int dldf, int cols)
{
  const int row = blockIdx.x;
  src += (size_t)row * sld;
  __shared__ float sb[256];
  float ss = 0.f;
  for (int c = threadIdx.x; c < cols; c += 256) { float v = src[c]; ss += v * v; }
  sb[threadIdx.x] = ss; __syncthreads();
  for (int s = 128; s > 0; s >>= 1) { if (threadIdx.x < s) sb[threadIdx.x] += sb[threadIdx.x + s]; __syncthreads(); }
  const float inv = rsqrtf(sb[0] / cols + 1e-6f);
  for (int c = threadIdx.x; c < cols; c += 256) {
    const float v = src[c] * inv * w[c];
    dstb[(size_t)row * dldb + c] = (__bf16)v;
    if (dstf) dstf[(size_t)row * dldf + c] = v;
  }
}

// ---------------- strided f32 -> bf16 convert (batched) ----------------
__global__ __launch_bounds__(256) void convert_bf16_kernel(
    const float* __restrict__ src, int sld, long sb,
    __bf16* __restrict__ dst, int dld, long db, int rows, int cols)
{
  src += (size_t)blockIdx.z * sb; dst += (size_t)blockIdx.z * db;
  const long idx = (long)blockIdx.x * 256 + threadIdx.x;
  if (idx >= (long)rows * cols) return;
  const int r = (int)(idx / cols), c = (int)(idx % cols);
  dst[(size_t)r * dld + c] = (__bf16)src[(size_t)r * sld + c];
}

// ---------------- transpose-convert f32(rows,cols) -> bf16(cols,rows), batched ----------------
__global__ __launch_bounds__(256) void transpose_bf16_kernel(
    const float* __restrict__ src, int sld, long sb,
    __bf16* __restrict__ dst, int dld, long db, int rows, int cols)
{
  src += (size_t)blockIdx.z * sb; dst += (size_t)blockIdx.z * db;
  const long idx = (long)blockIdx.x * 256 + threadIdx.x;
  if (idx >= (long)rows * cols) return;
  const int r = (int)(idx / cols), c = (int)(idx % cols);
  dst[(size_t)c * dld + r] = (__bf16)src[(size_t)r * sld + c];
}

// ---------------- RoPE: rotate k_pe and q_pe, write bf16 into concatenated attention operands ----------------
__global__ __launch_bounds__(512) void rope_kernel(
    const int* __restrict__ pos, const float* __restrict__ step1, const float* __restrict__ qf,
    __bf16* __restrict__ Abuf, __bf16* __restrict__ Bbuf)
{
  const int t = blockIdx.x;
  const int h = threadIdx.x >> 5, i = threadIdx.x & 31;
  const float p = (float)pos[t];
  const float freq = powf(10000.f, -((float)(2 * i)) / (float)DR_);
  const float ang = p * freq;
  const float cs = cosf(ang), sn = sinf(ang);
  // q_pe (head h): qf[t, h*192 + 128 .. h*192 + 191]
  const float* qp = qf + (size_t)t * QFN_ + h * (DN_ + DR_) + DN_;
  __bf16* ab = Abuf + (size_t)t * AW_ + h * A576_ + KVL_;
  const float x1 = qp[i], x2 = qp[i + 32];
  ab[i]      = (__bf16)(x1 * cs - x2 * sn);
  ab[i + 32] = (__bf16)(x1 * sn + x2 * cs);
  if (h == 0) {
    const float* kp = step1 + (size_t)t * STEP1N_ + (QL_ + KVL_);
    __bf16* bb = Bbuf + (size_t)t * A576_ + KVL_;
    const float y1 = kp[i], y2 = kp[i + 32];
    bb[i]      = (__bf16)(y1 * cs - y2 * sn);
    bb[i + 32] = (__bf16)(y1 * sn + y2 * cs);
  }
}

// ---------------- exact top-k threshold per row: radix binary search on order-preserving keys ----------------
__global__ __launch_bounds__(256) void topk_thr_kernel(
    const float* __restrict__ iscore, float* __restrict__ thr, int n, int topk)
{
  const int t = blockIdx.x;
  if (t < topk) { if (threadIdx.x == 0) thr[t] = -3.0e38f; return; }
  const float* row = iscore + (size_t)t * n;
  __shared__ int sb[256];
  unsigned lo = 0u, hi = 0xFFFFFFFFu;
  while (lo < hi) {
    const unsigned mid = (unsigned)(((unsigned long long)lo + hi + 1ull) >> 1);
    int cnt = 0;
    for (int s = threadIdx.x; s <= t; s += 256) {
      const unsigned b = __float_as_uint(row[s]);
      const unsigned key = (b & 0x80000000u) ? ~b : (b | 0x80000000u);
      cnt += (key >= mid) ? 1 : 0;
    }
    sb[threadIdx.x] = cnt; __syncthreads();
    for (int s2 = 128; s2 > 0; s2 >>= 1) { if (threadIdx.x < s2) sb[threadIdx.x] += sb[threadIdx.x + s2]; __syncthreads(); }
    const int total = sb[0];
    __syncthreads();
    if (total >= topk) lo = mid; else hi = mid - 1;
  }
  if (threadIdx.x == 0) {
    const unsigned b = (lo & 0x80000000u) ? (lo ^ 0x80000000u) : ~lo;
    thr[t] = __uint_as_float(b);
  }
}

// ---------------- row softmax f32 -> bf16 ----------------
__global__ __launch_bounds__(256) void softmax_kernel(
    const float* __restrict__ src, __bf16* __restrict__ dst, int n)
{
  const int t = blockIdx.x;
  src += (size_t)t * n; dst += (size_t)t * n;
  __shared__ float sb[256];
  float mx = -3.4e38f;
  for (int s = threadIdx.x; s < n; s += 256) mx = fmaxf(mx, src[s]);
  sb[threadIdx.x] = mx; __syncthreads();
  for (int s2 = 128; s2 > 0; s2 >>= 1) { if (threadIdx.x < s2) sb[threadIdx.x] = fmaxf(sb[threadIdx.x], sb[threadIdx.x + s2]); __syncthreads(); }
  mx = sb[0]; __syncthreads();
  float sum = 0.f;
  for (int s = threadIdx.x; s < n; s += 256) sum += expf(src[s] - mx);
  sb[threadIdx.x] = sum; __syncthreads();
  for (int s2 = 128; s2 > 0; s2 >>= 1) { if (threadIdx.x < s2) sb[threadIdx.x] += sb[threadIdx.x + s2]; __syncthreads(); }
  const float inv = 1.f / sb[0];
  for (int s = threadIdx.x; s < n; s += 256) dst[s] = (__bf16)(expf(src[s] - mx) * inv);
}

// ---------------- SwiGLU activation fuse ----------------
__global__ __launch_bounds__(256) void silu_mul_kernel(
    const float* __restrict__ g, const float* __restrict__ u, __bf16* __restrict__ o, long n)
{
  const long i = (long)blockIdx.x * 256 + threadIdx.x;
  if (i >= n) return;
  const float x = g[i];
  o[i] = (__bf16)((x / (1.f + expf(-x))) * u[i]);
}

// =====================================================================
extern "C" void kernel_launch(void* const* d_in, const int* in_sizes, int n_in,
                              void* d_out, int out_size, void* d_ws, size_t ws_size,
                              hipStream_t stream) {
  const int*   pos       = (const int*)d_in[0];
  const float* hs        = (const float*)d_in[1];
  const float* residual  = (const float*)d_in[2];
  const float* ln1_w     = (const float*)d_in[3];
  const float* ln2_w     = (const float*)d_in[4];
  const float* q_a_ln_w  = (const float*)d_in[5];
  const float* kv_a_ln_w = (const float*)d_in[6];
  const float* w_step1   = (const float*)d_in[7];
  const float* w_wq_b    = (const float*)d_in[8];
  const float* w_q_b     = (const float*)d_in[9];
  const float* w_uk      = (const float*)d_in[10];
  const float* w_uv      = (const float*)d_in[11];
  const float* w_o       = (const float*)d_in[12];
  const float* w_gate    = (const float*)d_in[13];
  const float* w_up      = (const float*)d_in[14];
  const float* w_down    = (const float*)d_in[15];

  float* out = (float*)d_out;                         // (T, HID)
  float* r2  = out + (size_t)T_ * HID_;               // (T, HID)

  size_t off = 0;
  auto take = [&](size_t bytes) -> char* {
    char* p = (char*)d_ws + off;
    off = (off + bytes + 255) & ~(size_t)255;
    return p;
  };

  float*  r1      = (float*)take((size_t)T_ * HID_ * 4);
  __bf16* h_bf    = (__bf16*)take((size_t)T_ * HID_ * 2);
  __bf16* wstep1b = (__bf16*)take((size_t)STEP1N_ * HID_ * 2);
  __bf16* wwqb_b  = (__bf16*)take((size_t)(IDXH_ * IDXD_) * QL_ * 2);
  __bf16* wqb_b   = (__bf16*)take((size_t)QFN_ * QL_ * 2);
  __bf16* wukT    = (__bf16*)take((size_t)NH_ * KVL_ * DN_ * 2);
  __bf16* wuvT    = (__bf16*)take((size_t)NH_ * DV_ * KVL_ * 2);
  __bf16* wo_b    = (__bf16*)take((size_t)HID_ * (NH_ * DV_) * 2);
  __bf16* wg_b    = (__bf16*)take((size_t)INTER_ * HID_ * 2);
  __bf16* wu_b    = (__bf16*)take((size_t)INTER_ * HID_ * 2);
  __bf16* wd_b    = (__bf16*)take((size_t)HID_ * INTER_ * 2);
  float*  step1   = (float*)take((size_t)T_ * STEP1N_ * 4);
  __bf16* qcn_b   = (__bf16*)take((size_t)T_ * QL_ * 2);
  float*  kv_f    = (float*)take((size_t)T_ * KVL_ * 4);
  __bf16* kvT     = (__bf16*)take((size_t)KVL_ * T_ * 2);
  __bf16* qidx_b  = (__bf16*)take((size_t)T_ * (IDXH_ * IDXD_) * 2);
  __bf16* idxk_b  = (__bf16*)take((size_t)T_ * IDXD_ * 2);
  float*  iscore  = (float*)take((size_t)T_ * T_ * 4);
  float*  thr     = (float*)take((size_t)T_ * 4);
  float*  qf      = (float*)take((size_t)T_ * QFN_ * 4);
  __bf16* qnope_b = (__bf16*)take((size_t)T_ * (NH_ * DN_) * 2);
  __bf16* Abuf    = (__bf16*)take((size_t)T_ * AW_ * 2);
  __bf16* Bbuf    = (__bf16*)take((size_t)T_ * A576_ * 2);
  float*  scores  = (float*)take((size_t)T_ * T_ * 4);
  __bf16* p_bf    = (__bf16*)take((size_t)T_ * T_ * 2);
  __bf16* olat_b  = (__bf16*)take((size_t)T_ * KVL_ * 2);
  __bf16* attn_b  = (__bf16*)take((size_t)T_ * (NH_ * DV_) * 2);
  float*  h2      = (float*)take((size_t)T_ * HID_ * 4);
  __bf16* h3_b    = (__bf16*)take((size_t)T_ * HID_ * 2);
  float*  gateb   = (float*)take((size_t)T_ * INTER_ * 4);
  float*  upb     = (float*)take((size_t)T_ * INTER_ * 4);
  __bf16* act_b   = (__bf16*)take((size_t)T_ * INTER_ * 2);

  auto gemm = [&](const __bf16* A, int lda, long sA, const __bf16* B, int ldb, long sB,
                  void* C, int ldc, long sC, int M, int N, int K, float alpha, int epi,
                  const float* a0, const float* a1, int as, int beta, int batch) {
    dim3 g((N + BN - 1) / BN, (M + BM - 1) / BM, batch);
    gemm_bf16_wmma<<<g, 256, 0, stream>>>(A, lda, sA, B, ldb, sB, C, ldc, sC,
                                          M, N, K, alpha, epi, a0, a1, as, beta);
  };
  auto convert = [&](const float* src, int sld, long sb, __bf16* dst, int dld, long db,
                     int rows, int cols, int batch) {
    long total = (long)rows * cols;
    dim3 g((unsigned)((total + 255) / 256), 1, batch);
    convert_bf16_kernel<<<g, 256, 0, stream>>>(src, sld, sb, dst, dld, db, rows, cols);
  };
  auto transpose = [&](const float* src, int sld, long sb, __bf16* dst, int dld, long db,
                       int rows, int cols, int batch) {
    long total = (long)rows * cols;
    dim3 g((unsigned)((total + 255) / 256), 1, batch);
    transpose_bf16_kernel<<<g, 256, 0, stream>>>(src, sld, sb, dst, dld, db, rows, cols);
  };

  // ---- 1. r1 = hs + residual ; h = rmsnorm(r1, ln1) (bf16) ----
  add_rmsnorm_kernel<<<T_, 256, 0, stream>>>(hs, residual, ln1_w, r1, h_bf, HID_);

  // ---- 2. weight casts to bf16 (WMMA operand precision) ----
  convert(w_step1, HID_, 0, wstep1b, HID_, 0, STEP1N_, HID_, 1);
  convert(w_wq_b, QL_, 0, wwqb_b, QL_, 0, IDXH_ * IDXD_, QL_, 1);
  convert(w_q_b, QL_, 0, wqb_b, QL_, 0, QFN_, QL_, 1);
  transpose(w_uk, KVL_, (long)DN_ * KVL_, wukT, DN_, (long)KVL_ * DN_, DN_, KVL_, NH_);
  transpose(w_uv, DV_, (long)KVL_ * DV_, wuvT, KVL_, (long)DV_ * KVL_, KVL_, DV_, NH_);
  convert(w_o, NH_ * DV_, 0, wo_b, NH_ * DV_, 0, HID_, NH_ * DV_, 1);
  convert(w_gate, HID_, 0, wg_b, HID_, 0, INTER_, HID_, 1);
  convert(w_up, HID_, 0, wu_b, HID_, 0, INTER_, HID_, 1);
  convert(w_down, INTER_, 0, wd_b, INTER_, 0, HID_, INTER_, 1);

  // ---- 3. step1 = h @ w_step1^T ----
  gemm(h_bf, HID_, 0, wstep1b, HID_, 0, step1, STEP1N_, 0,
       T_, STEP1N_, HID_, 1.f, EPI_F32, nullptr, nullptr, 0, 0, 1);

  // ---- 4. q/kv rmsnorms ----
  rmsnorm_kernel<<<T_, 256, 0, stream>>>(step1, STEP1N_, q_a_ln_w, qcn_b, QL_, nullptr, 0, QL_);
  rmsnorm_kernel<<<T_, 256, 0, stream>>>(step1 + QL_, STEP1N_, kv_a_ln_w, Bbuf, A576_, kv_f, KVL_, KVL_);
  transpose(kv_f, KVL_, 0, kvT, T_, 0, T_, KVL_, 1);  // kv^T for P@V GEMM

  // ---- 5. q_idx and qf projections ----
  gemm(qcn_b, QL_, 0, wwqb_b, QL_, 0, qidx_b, IDXH_ * IDXD_, 0,
       T_, IDXH_ * IDXD_, QL_, 1.f, EPI_BF16, nullptr, nullptr, 0, 0, 1);
  gemm(qcn_b, QL_, 0, wqb_b, QL_, 0, qf, QFN_, 0,
       T_, QFN_, QL_, 1.f, EPI_F32, nullptr, nullptr, 0, 0, 1);

  convert(step1 + QL_ + KVL_ + DR_, STEP1N_, 0, idxk_b, IDXD_, 0, T_, IDXD_, 1);
  convert(qf, QFN_, DN_ + DR_, qnope_b, NH_ * DN_, DN_, T_, DN_, NH_);  // per-head q_nope slices

  // ---- 6. RoPE -> concatenated [q_lat|q_pe_r] A operand and [kv_cn|k_pe_r] B operand ----
  rope_kernel<<<T_, 512, 0, stream>>>(pos, step1, qf, Abuf, Bbuf);

  // ---- 7. q_lat[h] = q_nope[h] @ w_uk[h] (batched over heads, bf16 out into Abuf) ----
  gemm(qnope_b, NH_ * DN_, DN_, wukT, DN_, (long)KVL_ * DN_, Abuf, AW_, A576_,
       T_, KVL_, DN_, 1.f, EPI_BF16, nullptr, nullptr, 0, 0, NH_);

  // ---- 8. indexer: iscore += relu(q_idx_h @ idx_k^T * d^-0.5) * idx_w[:,h] ----
  const float iscale = 0.08838834764831845f;  // 1/sqrt(128)
  for (int h = 0; h < IDXH_; ++h) {
    gemm(qidx_b + h * IDXD_, IDXH_ * IDXD_, 0, idxk_b, IDXD_, 0, iscore, T_, 0,
         T_, T_, IDXD_, iscale, EPI_IDX,
         step1 + (QL_ + KVL_ + DR_ + IDXD_) + h, nullptr, STEP1N_, (h ? 1 : 0), 1);
  }

  // ---- 9. top-k threshold per query row ----
  topk_thr_kernel<<<T_, 256, 0, stream>>>(iscore, thr, T_, TOPK_);

  // ---- 10. per-head attention ----
  const float ascale = 0.07216878364870323f;  // 1/sqrt(192)
  for (int h = 0; h < NH_; ++h) {
    gemm(Abuf + h * A576_, AW_, 0, Bbuf, A576_, 0, scores, T_, 0,
         T_, T_, A576_, ascale, EPI_SCORES, iscore, thr, T_, 0, 1);
    softmax_kernel<<<T_, 256, 0, stream>>>(scores, p_bf, T_);
    gemm(p_bf, T_, 0, kvT, T_, 0, olat_b, KVL_, 0,
         T_, KVL_, T_, 1.f, EPI_BF16, nullptr, nullptr, 0, 0, 1);
    gemm(olat_b, KVL_, 0, wuvT + (size_t)h * DV_ * KVL_, KVL_, 0, attn_b + h * DV_, NH_ * DV_, 0,
         T_, DV_, KVL_, 1.f, EPI_BF16, nullptr, nullptr, 0, 0, 1);
  }

  // ---- 11. output projection + residual + rmsnorm ----
  gemm(attn_b, NH_ * DV_, 0, wo_b, NH_ * DV_, 0, h2, HID_, 0,
       T_, HID_, NH_ * DV_, 1.f, EPI_F32, nullptr, nullptr, 0, 0, 1);
  add_rmsnorm_kernel<<<T_, 256, 0, stream>>>(h2, r1, ln2_w, r2, h3_b, HID_);

  // ---- 12. SwiGLU MLP ----
  gemm(h3_b, HID_, 0, wg_b, HID_, 0, gateb, INTER_, 0,
       T_, INTER_, HID_, 1.f, EPI_F32, nullptr, nullptr, 0, 0, 1);
  gemm(h3_b, HID_, 0, wu_b, HID_, 0, upb, INTER_, 0,
       T_, INTER_, HID_, 1.f, EPI_F32, nullptr, nullptr, 0, 0, 1);
  {
    long n = (long)T_ * INTER_;
    silu_mul_kernel<<<(unsigned)((n + 255) / 256), 256, 0, stream>>>(gateb, upb, act_b, n);
  }
  gemm(act_b, INTER_, 0, wd_b, INTER_, 0, out, HID_, 0,
       T_, HID_, INTER_, 1.f, EPI_F32, nullptr, nullptr, 0, 0, 1);
}
